// NeuralTuringMemo_70703751627105
// MI455X (gfx1250) — matrix operand order repaired and checked by reading, not compiled, not run on hardware
//
#include <hip/hip_runtime.h>

#define K_EMB   512
#define D_EMB   64
#define SPATIAL 16384          // 16*32*32
#define N_ROWS  131072         // 8 * SPATIAL
#define BM      256            // rows per block = 16 waves x 16 rows
#define NWAVES  16
#define ES_STRIDE  66          // k-major bf16 row stride (33 dwords, odd -> conflict free)
#define EST_STRIDE 514         // d-major bf16 row stride (257 dwords, odd -> conflict free)
#define PT_STRIDE  34
#define OS_STRIDE  66

typedef __attribute__((ext_vector_type(16))) __bf16       v16bf;
typedef __attribute__((ext_vector_type(8)))  float        v8f;
typedef __attribute__((ext_vector_type(8)))  unsigned int v8u;

__device__ __forceinline__ unsigned int f2bf(float f) {
  unsigned u = __builtin_bit_cast(unsigned, f);
  u += 0x7FFFu + ((u >> 16) & 1u);          // round-to-nearest-even
  return u >> 16;
}

__global__ __launch_bounds__(512) void vq_soft_wmma(
    const float* __restrict__ xin, const float* __restrict__ emb,
    float* __restrict__ out)
{
  __shared__ __align__(16) unsigned short Es [K_EMB * ES_STRIDE];        // 67584 B (k-major)
  __shared__ __align__(16) unsigned short EsT[D_EMB * EST_STRIDE];       // 65792 B (d-major)
  __shared__ __align__(16) float          EnS[K_EMB];                    //  2048 B
  __shared__ __align__(16) unsigned short Pt [NWAVES * 16 * PT_STRIDE];  // 17408 B
  __shared__ __align__(16) float          OutS[NWAVES * 16 * OS_STRIDE]; // 67584 B

  const int tid  = threadIdx.x;
  const int wave = tid >> 5;           // 0..15
  const int lane = tid & 31;

  const long long n0 = (long long)blockIdx.x * BM;     // 16384 % 256 == 0: no b-crossing
  const int bidx = (int)(n0 >> 14);
  const int rem0 = (int)(n0 & (SPATIAL - 1));

  // ---- stage embedding table: k-major AND d-major bf16 copies, one global pass ----
  for (int idx = tid; idx < K_EMB * D_EMB; idx += 512) {
    int k = idx >> 6, d = idx & 63;
    unsigned short bv = (unsigned short)f2bf(emb[idx]);
    Es [k * ES_STRIDE  + d] = bv;
    EsT[d * EST_STRIDE + k] = bv;
  }
  // ---- embedding squared norms (fp32) ----
  for (int k = tid; k < K_EMB; k += 512) {
    const float* e = emb + k * D_EMB;
    float s = 0.f;
    #pragma unroll
    for (int d = 0; d < D_EMB; ++d) { float v = e[d]; s += v * v; }
    EnS[k] = s;
  }

  // ---- per-lane WMMA fragment offset maps (ISA 7.12.2, 16-bit A 16x32 / B 32x16) ----
  const int halfSel = lane >> 4;
  const int nCol    = lane & 15;
  int koffA[8], koffB[8];
  #pragma unroll
  for (int j = 0; j < 8; ++j) {
    koffA[j] = ((j >> 2) << 4) + (halfSel << 3) + ((j & 3) << 1);
    koffB[j] = ((j >> 2) << 4) + (halfSel << 3) + ((j & 3) << 1);
  }

  // ---- persistent A fragments of X: direct coalesced global loads (X read once) ----
  // For fixed j: lanes 0-15 read 16 consecutive spatial floats at channel d (one 64B
  // line), lanes 16-31 the same at channel d+8 (second full line).
  const float* xg = xin + (long long)bidx * (64LL * SPATIAL) + rem0 + wave * 16 + nCol;
  v8u ax0u, ax1u;
  #pragma unroll
  for (int j = 0; j < 8; ++j) {
    const int d0 = koffA[j];
    ax0u[j] = f2bf(xg[(long long)d0 * SPATIAL])
            | (f2bf(xg[(long long)(d0 + 1) * SPATIAL]) << 16);
    ax1u[j] = f2bf(xg[(long long)(d0 + 32) * SPATIAL])
            | (f2bf(xg[(long long)(d0 + 33) * SPATIAL]) << 16);
  }
  const v16bf ax0 = __builtin_bit_cast(v16bf, ax0u);
  const v16bf ax1 = __builtin_bit_cast(v16bf, ax1u);

  __syncthreads();

  unsigned short* ptw = &Pt[wave * 16 * PT_STRIDE];

  v8f acc[4];
  #pragma unroll
  for (int t = 0; t < 4; ++t) acc[t] = {};
  v8f lsum = {};

  for (int pair = 0; pair < K_EMB / 32; ++pair) {
    const int k0p = pair * 32;
    // -- scores for two 16-embedding chunks: s = ||e||^2 - 2 x.e  (||x||^2 dropped) --
    #pragma unroll
    for (int half = 0; half < 2; ++half) {
      const int k0 = k0p + half * 16;
      const unsigned short* brow = &Es[(k0 + nCol) * ES_STRIDE];
      v8u b0u, b1u;
      #pragma unroll
      for (int j = 0; j < 8; ++j) {
        b0u[j] = *(const unsigned int*)&brow[koffB[j]];        // d = 0..31
        b1u[j] = *(const unsigned int*)&brow[32 + koffB[j]];   // d = 32..63
      }
      v8f c = {};
      c = __builtin_amdgcn_wmma_f32_16x16x32_bf16(false, ax0, false,
              __builtin_bit_cast(v16bf, b0u), (short)0, c, false, false);
      c = __builtin_amdgcn_wmma_f32_16x16x32_bf16(false, ax1, false,
              __builtin_bit_cast(v16bf, b1u), (short)0, c, false, false);
      const float en = EnS[k0 + nCol];
      #pragma unroll
      for (int v = 0; v < 8; ++v) {
        float p = __expf(en - 2.f * c[v]);
        lsum[v] += p;
        // C/D-layout (M=v+8*halfSel, N=nCol) -> A-layout via per-wave LDS tile
        ptw[(v + (halfSel << 3)) * PT_STRIDE + half * 16 + nCol] = (unsigned short)f2bf(p);
      }
    }
    // -- A fragment of P (16x32 bf16) --
    v8u apu;
    #pragma unroll
    for (int j = 0; j < 8; ++j)
      apu[j] = *(const unsigned int*)&ptw[nCol * PT_STRIDE + koffA[j]];
    const v16bf ap = __builtin_bit_cast(v16bf, apu);
    // -- acc(16x64) += P(16x32) * E_chunk(32x64): B pairs along k from d-major EsT --
    #pragma unroll
    for (int t = 0; t < 4; ++t) {
      const unsigned short* etrow = &EsT[(t * 16 + nCol) * EST_STRIDE + k0p];
      v8u bu;
      #pragma unroll
      for (int j = 0; j < 8; ++j)
        bu[j] = *(const unsigned int*)&etrow[koffB[j]];
      acc[t] = __builtin_amdgcn_wmma_f32_16x16x32_bf16(false, ap, false,
                   __builtin_bit_cast(v16bf, bu), (short)0, acc[t], false, false);
    }
  }

  // ---- softmax denominators: sum each row's 16 partials across its half-wave ----
  v8f rinv;
  #pragma unroll
  for (int v = 0; v < 8; ++v) {
    float s = lsum[v];
    s += __shfl_xor(s, 1, 32);
    s += __shfl_xor(s, 2, 32);
    s += __shfl_xor(s, 4, 32);
    s += __shfl_xor(s, 8, 32);   // masks < 16: stays within the 16-lane row group
    rinv[v] = 1.0f / s;
  }

  // ---- normalize + stage (16 rows x 64 d, fp32) in this wave's LDS slab ----
  float* osw = &OutS[wave * 16 * OS_STRIDE];
  #pragma unroll
  for (int t = 0; t < 4; ++t)
    #pragma unroll
    for (int v = 0; v < 8; ++v)
      osw[(v + (halfSel << 3)) * OS_STRIDE + t * 16 + nCol] = acc[t][v] * rinv[v];

  // ---- coalesced store back to (B, D, T, H, W): 64B float4 runs along spatial ----
  float* obase = out + (long long)bidx * (64LL * SPATIAL) + rem0 + wave * 16;
  #pragma unroll
  for (int dd = 0; dd < 2; ++dd) {
    const int d = lane + dd * 32;
    float* dst = obase + (long long)d * SPATIAL;
    #pragma unroll
    for (int q = 0; q < 4; ++q) {
      float4 vv;
      vv.x = osw[(q * 4 + 0) * OS_STRIDE + d];
      vv.y = osw[(q * 4 + 1) * OS_STRIDE + d];
      vv.z = osw[(q * 4 + 2) * OS_STRIDE + d];
      vv.w = osw[(q * 4 + 3) * OS_STRIDE + d];
      *(float4*)(dst + q * 4) = vv;
    }
  }
}

extern "C" void kernel_launch(void* const* d_in, const int* in_sizes, int n_in,
                              void* d_out, int out_size, void* d_ws, size_t ws_size,
                              hipStream_t stream) {
  (void)in_sizes; (void)n_in; (void)out_size; (void)d_ws; (void)ws_size;
  const float* xin = (const float*)d_in[0];   // (8, 64, 16, 32, 32) fp32
  const float* emb = (const float*)d_in[1];   // (512, 64) fp32
  float* out = (float*)d_out;                 // (8, 64, 16, 32, 32) fp32
  vq_soft_wmma<<<dim3(N_ROWS / BM), dim3(512), 0, stream>>>(xin, emb, out);
}
